// GptOssAttention_4922032521855
// MI455X (gfx1250) — matrix operand order repaired and checked by reading, not compiled, not run on hardware
//
#include <hip/hip_runtime.h>
#include <hip/hip_bf16.h>

// ---------------------------------------------------------------------------
// CDNA5 (gfx1250) WMMA types
// ---------------------------------------------------------------------------
typedef __attribute__((ext_vector_type(16))) __bf16 v16bf;
typedef __attribute__((ext_vector_type(8)))  float  v8f;

// Problem constants (from the reference)
#define T_TOK   2048
#define HID_C   2880
#define HQ_C    64
#define HKV_C   8
#define D_C     64
#define QSZ     4096      // HQ*D
#define KVSZ    512       // HKV*D
#define S_LEN   1024
#define WIN     128
#define NKEY    144       // 16 + WIN keys covered by a 16-token tile
#define NKEYP   160       // padded to multiple of 32 for the PV WMMA

// ---------------------------------------------------------------------------
// CDNA5 async global->LDS copy (ASYNCcnt-tracked, no VGPR data path).
// VGLOBAL encoding: VDST = per-lane LDS byte address, VADDR = 64-bit address.
// Generic pointers to __shared__ carry the LDS byte offset in their low 32b.
// ---------------------------------------------------------------------------
__device__ inline void async_copy_b128(const void* g, void* lds) {
    const unsigned lo = (unsigned)(unsigned long long)lds;
    asm volatile("global_load_async_to_lds_b128 %0, %1, off"
                 :: "v"(lo), "v"((unsigned long long)g) : "memory");
}
__device__ inline void async_wait0() {
    asm volatile("s_wait_asynccnt 0" ::: "memory");
}

// ---------------------------------------------------------------------------
// Fragment loaders (per CDNA5 ISA 7.12.2 VGPR layouts, wave32)
// ---------------------------------------------------------------------------
// A matrix 16x32 (MxK), source row-major: base[row*ld + k], ld even.
// lanes 0-15: M=lane, V0-3 = K0..7 pairs, V4-7 = K16..23 pairs
// lanes 16-31: M=lane-16, V0-3 = K8..15, V4-7 = K24..31
__device__ inline v16bf load_frag_a(const __bf16* base, int ld) {
    const int l = threadIdx.x & 31;
    const int row = l & 15;
    const int khalf = (l & 16) ? 8 : 0;
    union { unsigned u[8]; v16bf v; } r;
    const unsigned short* p = (const unsigned short*)(base + (long)row * ld);
#pragma unroll
    for (int i = 0; i < 8; ++i) {
        const int k = ((i & 4) ? 16 : 0) + khalf + ((i & 3) << 1);
        r.u[i] = *(const unsigned*)(p + k);
    }
    return r.v;
}

// B matrix 32x16 (KxN), source K-contiguous per column: base[col*ld + k].
// lanes 0-15: N=lane, V0-3 = K0..7 pairs, V4-7 = K16..23; lanes 16-31: K8..15, K24..31
__device__ inline v16bf load_frag_b(const __bf16* base, int ld) {
    const int l = threadIdx.x & 31;
    const int col = l & 15;
    const int kbase = (l & 16) ? 8 : 0;
    union { unsigned u[8]; v16bf v; } r;
    const unsigned short* p = (const unsigned short*)(base + (long)col * ld);
#pragma unroll
    for (int i = 0; i < 8; ++i) {
        r.u[i] = *(const unsigned*)(p + ((i & 4) ? 16 : 0) + kbase + ((i & 3) << 1));
    }
    return r.v;
}

__device__ inline v8f wmma_bf16(v16bf a, v16bf b, v8f c) {
    return __builtin_amdgcn_wmma_f32_16x16x32_bf16(false, a, false, b,
                                                   (short)0, c, false, false);
}

// ---------------------------------------------------------------------------
// f32 -> bf16 conversion, 4-wide
// ---------------------------------------------------------------------------
__global__ __launch_bounds__(256) void cvt4_kernel(const float* __restrict__ src,
                                                   __bf16* __restrict__ dst,
                                                   long n4) {
    long i = (long)blockIdx.x * blockDim.x + threadIdx.x;
    const long stride = (long)gridDim.x * blockDim.x;
    for (; i < n4; i += stride) {
        float4 f = ((const float4*)src)[i];
        union { __bf16 h[4]; unsigned long long u; } r;
        r.h[0] = (__bf16)f.x; r.h[1] = (__bf16)f.y;
        r.h[2] = (__bf16)f.z; r.h[3] = (__bf16)f.w;
        ((unsigned long long*)dst)[i] = r.u;
    }
}

// ---------------------------------------------------------------------------
// GEMM + bias: C[MxN] = A[MxK](bf16) * B[KxN](bf16) + bias, C f32.
// Block tile 128xBN, 256 threads = 8 waves, each wave a 16xBN strip.
// M % 128 == 0, N % BN == 0, K % 32 == 0 (true for all our shapes).
// A tile is staged with CDNA5 async global->LDS copies; B tile goes through
// VGPRs because it is transposed on the LDS store.
// ---------------------------------------------------------------------------
#define GM_LDA 40   // 32 + 8 pad, keeps b128 LDS accesses 16B aligned
#define GM_LDB 40

template <int BN>
__global__ __launch_bounds__(256) void gemm_bias_kernel(
        const __bf16* __restrict__ A, const __bf16* __restrict__ B,
        const float* __restrict__ bias, float* __restrict__ C,
        int M, int N, int K) {
    constexpr int NSUB = BN / 16;
    constexpr int SEGS = BN / 8;   // 8-element segments per B row
    __shared__ __align__(16) __bf16 sA[128 * GM_LDA];
    __shared__ __align__(16) __bf16 sBt[BN * GM_LDB];   // [n][k], k-contig

    const int tid  = threadIdx.x;
    const int wave = tid >> 5;
    const int n0   = blockIdx.x * BN;
    const int m0   = blockIdx.y * 128;

    v8f acc[NSUB] = {};

    const int arow = tid >> 1, ahalf = tid & 1;

    for (int k0 = 0; k0 < K; k0 += 32) {
        // Stage A tile (128x32) via async global->LDS b128 copies
        const __bf16* gA = A + (long)(m0 + arow) * K + k0 + ahalf * 16;
        __bf16* lA = sA + arow * GM_LDA + ahalf * 16;
        async_copy_b128(gA,     lA);
        async_copy_b128(gA + 8, lA + 8);

        // Stage B tile (32xBN) transposed -> sBt[n][k]
#pragma unroll
        for (int sidx = tid; sidx < 4 * BN; sidx += 256) {
            const int kk = sidx / SEGS;
            const int nseg = (sidx % SEGS) * 8;
            union { uint4 u; __bf16 h[8]; } bvv;
            bvv.u = *(const uint4*)(B + (long)(k0 + kk) * N + n0 + nseg);
#pragma unroll
            for (int j = 0; j < 8; ++j) sBt[(nseg + j) * GM_LDB + kk] = bvv.h[j];
            if (k0 + 32 < K)
                __builtin_prefetch(B + (long)(k0 + 32 + kk) * N + n0 + nseg, 0, 1);
        }
        async_wait0();
        __syncthreads();

        v16bf af = load_frag_a(sA + wave * 16 * GM_LDA, GM_LDA);
        // software-pipelined B-fragment loads
        v16bf bcur = load_frag_b(sBt, GM_LDB);
#pragma unroll
        for (int s = 0; s < NSUB; ++s) {
            v16bf bnext;
            if (s + 1 < NSUB) bnext = load_frag_b(sBt + (s + 1) * 16 * GM_LDB, GM_LDB);
            acc[s] = wmma_bf16(af, bcur, acc[s]);
            bcur = bnext;
        }
        __syncthreads();
    }

    // Epilogue: C layout lane l -> N = l&15, M = r + 8*(l>>4)
    const int l = tid & 31;
    const int nn = n0 + (l & 15);
    const int mh = (l >> 4) * 8;
#pragma unroll
    for (int s = 0; s < NSUB; ++s) {
        const float bb = bias[nn + s * 16];
#pragma unroll
        for (int r = 0; r < 8; ++r) {
            C[(long)(m0 + wave * 16 + mh + r) * N + nn + s * 16] = acc[s][r] + bb;
        }
    }
}

// ---------------------------------------------------------------------------
// NeoX RoPE fused with f32 -> bf16 conversion (theta = 150000, D = 64)
// ---------------------------------------------------------------------------
__global__ __launch_bounds__(256) void rope_kernel(const float* __restrict__ src,
                                                   const int* __restrict__ pos,
                                                   __bf16* __restrict__ dst,
                                                   int H) {
    const int idx = blockIdx.x * 256 + threadIdx.x;
    const int total = T_TOK * H * 32;
    if (idx >= total) return;
    const int i = idx & 31;
    const int th = idx >> 5;
    const int h = th % H;
    const int t = th / H;
    const float p = (float)pos[t];
    // inv_freq = theta^(-2i/64) = exp(-i * ln(theta)/32)
    const float inv = __expf((float)i * -0.37244971f);
    const float ang = p * inv;
    float sn, cs;
    __sincosf(ang, &sn, &cs);
    const float* x = src + ((long)t * H + h) * 64;
    __bf16* o = dst + ((long)t * H + h) * 64;
    const float x1 = x[i], x2 = x[i + 32];
    o[i]      = (__bf16)(x1 * cs - x2 * sn);
    o[i + 32] = (__bf16)(x1 * sn + x2 * cs);
}

// ---------------------------------------------------------------------------
// Sliding-window GQA attention (W=128, D=64, G=8 q-heads per kv-head).
// Grid: (T/16 token tiles, HKV). Block: 128 threads = 4 waves.
// Wave w handles q group members {w, w+4} sequentially.
// ---------------------------------------------------------------------------
#define AT_LDK 72    // 64 + 8 pad  (sK[key][dim], dim-contiguous)
#define AT_LDV 160   // sVt[dim][key], key-contiguous, padded to 160 keys
#define AT_LDP 160   // sP[m][key], key-contiguous, padded

__global__ __launch_bounds__(128) void attn_kernel(const __bf16* __restrict__ qb,
                                                   const __bf16* __restrict__ kb,
                                                   const __bf16* __restrict__ vb,
                                                   __bf16* __restrict__ ob) {
    __shared__ __align__(16) __bf16 sK[NKEY * AT_LDK];
    __shared__ __align__(16) __bf16 sVt[D_C * AT_LDV];
    __shared__ __align__(16) __bf16 sP[4 * 16 * AT_LDP];

    const int tid  = threadIdx.x;
    const int wave = tid >> 5;
    const int l    = tid & 31;
    const int tile = blockIdx.x;
    const int kvh  = blockIdx.y;
    const int bsel = (tile * 16) / S_LEN;
    const int s0   = (tile * 16) % S_LEN;
    const long tok0 = (long)bsel * S_LEN + s0;

    // --- stage K window (async, row-major) and V window (transposed) ---
    for (int idx = tid; idx < NKEY * 8; idx += 128) {
        const int row = idx >> 3;
        const int seg = (idx & 7) * 8;
        const int sk = s0 - WIN + row;
        union { uint4 u; __bf16 h[8]; } vv;
        if (sk >= 0) {
            const long koff = ((long)bsel * S_LEN + sk) * KVSZ + kvh * 64 + seg;
            async_copy_b128(kb + koff, sK + row * AT_LDK + seg);
            vv.u = *(const uint4*)(vb + koff);
        } else {
            *(uint4*)(sK + row * AT_LDK + seg) = make_uint4(0u, 0u, 0u, 0u);
            vv.u = make_uint4(0u, 0u, 0u, 0u);
        }
#pragma unroll
        for (int j = 0; j < 8; ++j) sVt[(seg + j) * AT_LDV + row] = vv.h[j];
    }
    // zero the padded key columns of Vt (keys 144..159)
    for (int idx = tid; idx < D_C * (NKEYP - NKEY); idx += 128) {
        sVt[(idx >> 4) * AT_LDV + NKEY + (idx & 15)] = (__bf16)0.0f;
    }
    async_wait0();
    __syncthreads();

    const int mh = (l >> 4) * 8;   // M offset of this lane's accumulator rows
    const int nb = l & 15;         // N (column) of this lane
    __bf16* pw = sP + wave * 16 * AT_LDP;

#pragma unroll 1
    for (int hh = 0; hh < 2; ++hh) {
        const int g = wave + hh * 4;
        const int head = kvh * 8 + g;
        const __bf16* qbase = qb + tok0 * QSZ + head * 64;

        // Q fragments: 16 tokens x 64 dims, two K-chunks of 32
        v16bf qf0 = load_frag_a(qbase, QSZ);
        v16bf qf1 = load_frag_a(qbase + 32, QSZ);

        // scores = q @ k^T : 9 blocks of 16 keys
        v8f sc[9] = {};
#pragma unroll
        for (int j = 0; j < 9; ++j) {
            v16bf b0 = load_frag_b(sK + j * 16 * AT_LDK, AT_LDK);
            sc[j] = wmma_bf16(qf0, b0, sc[j]);
            v16bf b1 = load_frag_b(sK + j * 16 * AT_LDK + 32, AT_LDK);
            sc[j] = wmma_bf16(qf1, b1, sc[j]);
        }

        // masked softmax over the 144 keys (row-wise; row values live in the
        // 16 lanes of each half-wave x 9 register blocks)
#pragma unroll
        for (int r = 0; r < 8; ++r) {
            const int m = mh + r;
            float vals[9];
            float rmax = -1e30f;
#pragma unroll
            for (int j = 0; j < 9; ++j) {
                const int jn = j * 16 + nb;
                const bool valid = (jn >= m) && (jn <= m + WIN) && (s0 - WIN + jn >= 0);
                const float x = valid ? sc[j][r] * 0.125f : -1e30f;
                vals[j] = x;
                rmax = fmaxf(rmax, x);
            }
            rmax = fmaxf(rmax, __shfl_xor(rmax, 1));
            rmax = fmaxf(rmax, __shfl_xor(rmax, 2));
            rmax = fmaxf(rmax, __shfl_xor(rmax, 4));
            rmax = fmaxf(rmax, __shfl_xor(rmax, 8));
            float rsum = 0.0f;
#pragma unroll
            for (int j = 0; j < 9; ++j) { vals[j] = __expf(vals[j] - rmax); rsum += vals[j]; }
            rsum += __shfl_xor(rsum, 1);
            rsum += __shfl_xor(rsum, 2);
            rsum += __shfl_xor(rsum, 4);
            rsum += __shfl_xor(rsum, 8);
            const float rinv = 1.0f / rsum;
#pragma unroll
            for (int j = 0; j < 9; ++j) {
                pw[m * AT_LDP + j * 16 + nb] = (__bf16)(vals[j] * rinv);
            }
            pw[m * AT_LDP + NKEY + nb] = (__bf16)0.0f;   // padded keys
        }
        __syncthreads();

        // o = p @ v : K = 160 (5 chunks of 32), N = 64 (4 subtiles)
        v8f ov[4] = {};
#pragma unroll
        for (int kc = 0; kc < 5; ++kc) {
            v16bf pf = load_frag_a(pw + kc * 32, AT_LDP);
#pragma unroll
            for (int s = 0; s < 4; ++s) {
                v16bf vf = load_frag_b(sVt + s * 16 * AT_LDV + kc * 32, AT_LDV);
                ov[s] = wmma_bf16(pf, vf, ov[s]);
            }
        }

        __bf16* obase = ob + tok0 * QSZ + head * 64;
#pragma unroll
        for (int s = 0; s < 4; ++s) {
#pragma unroll
            for (int r = 0; r < 8; ++r) {
                obase[(long)(mh + r) * QSZ + s * 16 + nb] = (__bf16)ov[s][r];
            }
        }
        __syncthreads();
    }
}

// ---------------------------------------------------------------------------
// Host-side orchestration
// ---------------------------------------------------------------------------
extern "C" void kernel_launch(void* const* d_in, const int* in_sizes, int n_in,
                              void* d_out, int out_size, void* d_ws, size_t ws_size,
                              hipStream_t stream) {
    (void)in_sizes; (void)n_in; (void)out_size; (void)ws_size;
    const float* h   = (const float*)d_in[0];
    const int*   pos = (const int*)d_in[1];
    const float* Wq  = (const float*)d_in[2];
    const float* bq  = (const float*)d_in[3];
    const float* Wk  = (const float*)d_in[4];
    const float* bk  = (const float*)d_in[5];
    const float* Wv  = (const float*)d_in[6];
    const float* bv  = (const float*)d_in[7];
    const float* Wo  = (const float*)d_in[8];
    const float* bo  = (const float*)d_in[9];
    float* out = (float*)d_out;
    char*  ws  = (char*)d_ws;

    // workspace layout (all offsets multiples of 256)
    const size_t SZ_HBF  = (size_t)T_TOK * HID_C * 2;   // 11,796,480
    const size_t SZ_WQB  = (size_t)HID_C * QSZ * 2;     // 23,592,960
    const size_t SZ_WKB  = (size_t)HID_C * KVSZ * 2;    //  2,949,120
    const size_t SZ_WOB  = (size_t)QSZ * HID_C * 2;     // 23,592,960
    const size_t SZ_QF   = (size_t)T_TOK * QSZ * 4;     // 33,554,432
    const size_t SZ_KF   = (size_t)T_TOK * KVSZ * 4;    //  4,194,304
    const size_t SZ_QB   = (size_t)T_TOK * QSZ * 2;     // 16,777,216
    const size_t SZ_KB   = (size_t)T_TOK * KVSZ * 2;    //  2,097,152

    size_t off = 0;
    __bf16* h_bf  = (__bf16*)(ws + off); off += SZ_HBF;
    __bf16* Wq_bf = (__bf16*)(ws + off); off += SZ_WQB;
    __bf16* Wk_bf = (__bf16*)(ws + off); off += SZ_WKB;
    __bf16* Wv_bf = (__bf16*)(ws + off); off += SZ_WKB;
    __bf16* Wo_bf = (__bf16*)(ws + off); off += SZ_WOB;
    float*  q_f   = (float*) (ws + off);
    __bf16* ao_bf = (__bf16*)(ws + off); off += SZ_QF;   // ao aliases q_f (dead after rope)
    float*  k_f   = (float*) (ws + off); off += SZ_KF;
    float*  v_f   = (float*) (ws + off); off += SZ_KF;
    __bf16* q_bf  = (__bf16*)(ws + off); off += SZ_QB;
    __bf16* k_bf  = (__bf16*)(ws + off); off += SZ_KB;
    __bf16* v_bf  = (__bf16*)(ws + off); off += SZ_KB;

    // 1) convert hidden + weights to bf16
    cvt4_kernel<<<2048, 256, 0, stream>>>(h,  h_bf,  (long)T_TOK * HID_C / 4);
    cvt4_kernel<<<2048, 256, 0, stream>>>(Wq, Wq_bf, (long)HID_C * QSZ / 4);
    cvt4_kernel<<<1024, 256, 0, stream>>>(Wk, Wk_bf, (long)HID_C * KVSZ / 4);
    cvt4_kernel<<<1024, 256, 0, stream>>>(Wv, Wv_bf, (long)HID_C * KVSZ / 4);
    cvt4_kernel<<<2048, 256, 0, stream>>>(Wo, Wo_bf, (long)QSZ * HID_C / 4);

    // 2) QKV projections (WMMA GEMM + bias, f32 accumulate)
    gemm_bias_kernel<128><<<dim3(QSZ / 128,  T_TOK / 128), 256, 0, stream>>>(
        h_bf, Wq_bf, bq, q_f, T_TOK, QSZ, HID_C);
    gemm_bias_kernel<128><<<dim3(KVSZ / 128, T_TOK / 128), 256, 0, stream>>>(
        h_bf, Wk_bf, bk, k_f, T_TOK, KVSZ, HID_C);
    gemm_bias_kernel<128><<<dim3(KVSZ / 128, T_TOK / 128), 256, 0, stream>>>(
        h_bf, Wv_bf, bv, v_f, T_TOK, KVSZ, HID_C);

    // 3) RoPE (q, k) + plain convert (v) into bf16
    rope_kernel<<<(T_TOK * HQ_C * 32 + 255) / 256, 256, 0, stream>>>(q_f, pos, q_bf, HQ_C);
    rope_kernel<<<(T_TOK * HKV_C * 32 + 255) / 256, 256, 0, stream>>>(k_f, pos, k_bf, HKV_C);
    cvt4_kernel<<<1024, 256, 0, stream>>>(v_f, v_bf, (long)T_TOK * KVSZ / 4);

    // 4) sliding-window GQA attention
    attn_kernel<<<dim3(T_TOK / 16, HKV_C), 128, 0, stream>>>(q_bf, k_bf, v_bf, ao_bf);

    // 5) output projection -> d_out (f32); N=2880 needs the 64-wide tile
    gemm_bias_kernel<64><<<dim3(HID_C / 64, T_TOK / 128), 256, 0, stream>>>(
        ao_bf, Wo_bf, bo, out, T_TOK, HID_C, QSZ);
}
